// CustomAutoencoder_72189810311776
// MI455X (gfx1250) — compile-verified, matrix operand order checked by reading, not run on hardware
//
#include <hip/hip_runtime.h>
#include <hip/hip_bf16.h>

typedef __bf16 bf16;
typedef __attribute__((ext_vector_type(16))) __bf16 v16bf;
typedef __attribute__((ext_vector_type(8)))  __bf16 v8bf;
typedef __attribute__((ext_vector_type(4)))  __bf16 v4bf;
typedef __attribute__((ext_vector_type(8)))  float  v8f;
typedef __attribute__((ext_vector_type(4)))  unsigned int v4u;
typedef __attribute__((ext_vector_type(8)))  int v8i_;
typedef __attribute__((ext_vector_type(4)))  int v4i_;

union ABFrag { v16bf v; v8bf h[2]; };

enum : int {
  Bdim = 8192, Sdim = 4096,
  H1 = 196, H1P = 224,      // pad to multiple of 32
  H2 = 10,  H2P = 32,
  D4 = 1024, LAT = 32, DD1 = 1024, DD2 = 2048
};

#define BM 128
#define BN 128
#define BK 32

// ---------------------------------------------------------------------------
// Convert fp32 (optionally elementwise-masked) weights/activations to bf16,
// zero-padding to [rowsP, colsP].
// ---------------------------------------------------------------------------
__global__ void cvt_pad_bf16(const float* __restrict__ src,
                             const float* __restrict__ mask,
                             bf16* __restrict__ dst,
                             int rows, int cols, int rowsP, int colsP)
{
  size_t idx = (size_t)blockIdx.x * blockDim.x + threadIdx.x;
  size_t total = (size_t)rowsP * colsP;
  if (idx >= total) return;
  int r = (int)(idx / colsP);
  int c = (int)(idx % colsP);
  float v = 0.0f;
  if (r < rows && c < cols) {
    v = src[(size_t)r * cols + c];
    if (mask) v *= mask[(size_t)r * cols + c];
  }
  dst[idx] = (bf16)v;
}

// ---------------------------------------------------------------------------
// TDM: DMA one [128 rows x 32 cols] bf16 tile (row stride K elements) from
// global memory into LDS at lds_off.  D# layout per CDNA5 ISA ch.8.
// Issued by one wave; completion tracked by TENSORcnt.
// This toolchain exposes the 6-arg builtin:
//   (uint32x4 g0, int32x8 g1, int32x4 g2, int32x4 g3, int32x8 g4, i32 cpol)
// ---------------------------------------------------------------------------
__device__ __forceinline__ void tdm_load_tile(const bf16* gsrc, unsigned lds_off,
                                              int rowStrideElems)
{
  unsigned long long ga = (unsigned long long)(size_t)gsrc;

  // ---- D# group 0 (128b): count | lds_addr | global_addr[56:0] | type=2
  v4u g0;
  g0[0] = 1u;                                        // count=1, user desc
  g0[1] = lds_off;                                   // lds_addr (bytes)
  g0[2] = (unsigned)(ga & 0xFFFFFFFFu);              // global_addr[31:0]
  g0[3] = (unsigned)((ga >> 32) & 0x01FFFFFFu)       // global_addr[56:32]
        | (2u << 30);                                // type = 2 ("image")

  // ---- D# group 1 (256b)
  unsigned td0 = (unsigned)rowStrideElems;           // tensor_dim0 (elements)
  unsigned td1 = 0x00FFFFFFu;                        // tensor_dim1 (large, in-bounds by construction)
  unsigned long long s0 = (unsigned long long)rowStrideElems; // dim0 stride (elements)
  v8i_ g1;
  g1[0] = (int)(1u << 16);                           // data_size=1 (2 bytes); wg_mask=0
  g1[1] = (int)((td0 & 0xFFFFu) << 16);              // [31:16]=tensor_dim0[15:0]
  g1[2] = (int)(((td0 >> 16) & 0xFFFFu) | ((td1 & 0xFFFFu) << 16));
  g1[3] = (int)(((td1 >> 16) & 0xFFFFu) | ((unsigned)BK << 16)); // tile_dim0=32
  g1[4] = (int)((unsigned)BM);                       // tile_dim1=128, tile_dim2=0
  g1[5] = (int)(s0 & 0xFFFFFFFFu);                   // tensor_dim0_stride[31:0]
  g1[6] = (int)((s0 >> 32) & 0xFFFFu);               // stride[47:32]; dim1_stride[15:0]=0
  g1[7] = 0;                                         // dim1_stride[47:16]=0

  v4i_ g2 = {0, 0, 0, 0};                            // unused (<=2D tensor)
  v4i_ g3 = {0, 0, 0, 0};
  v8i_ g4 = {0, 0, 0, 0, 0, 0, 0, 0};                // unused trailing group

  __builtin_amdgcn_tensor_load_to_lds(g0, g1, g2, g3, g4, 0);
}

// ---------------------------------------------------------------------------
// GEMM: out[M,Np] = act(A[M,K] @ W[K,Np] + bias)   (bf16 in, fp32 accumulate)
// ACT: 0 = relu, 1 = sigmoid.  OutT: bf16 (intermediate) or float (final).
// M multiple of 128, K multiple of 32. Np arbitrary (bounds-checked).
// A tile staged via Tensor Data Mover (async DMA, TENSORcnt);
// W tile staged via VGPR loads with in-register transpose to [n][k].
// ---------------------------------------------------------------------------
template<int ACT, typename OutT>
__global__ void __launch_bounds__(256)
gemm_bias_act(const bf16* __restrict__ A, const bf16* __restrict__ W,
              const float* __restrict__ bias, OutT* __restrict__ out,
              int M, int K, int Np, int Nbias)
{
  __shared__ __align__(16) bf16 As[2][BM * BK];   // [m][k] row-major (TDM dest)
  __shared__ __align__(16) bf16 Bs[2][BN * BK];   // [n][k] transposed

  const int tid  = threadIdx.x;
  const int lane = tid & 31;
  const int wave = tid >> 5;        // 8 waves
  const int wm   = wave >> 1;       // 0..3 -> M offset wm*32
  const int wn   = wave & 1;        // 0..1 -> N offset wn*64
  const int bm   = blockIdx.x * BM;
  const int bn   = blockIdx.y * BN;

  v8f acc[2][4];
#pragma unroll
  for (int i = 0; i < 2; ++i)
#pragma unroll
    for (int j = 0; j < 4; ++j)
#pragma unroll
      for (int e = 0; e < 8; ++e) acc[i][j][e] = 0.0f;

  auto tdmA = [&](int kt, int buf) {
    if (wave == 0)
      tdm_load_tile(A + (size_t)bm * K + kt,
                    (unsigned)(size_t)&As[buf][0], K);
  };

  auto loadB = [&](int kt, int buf) {
#pragma unroll
    for (int it = 0; it < 4; ++it) {
      int k = (tid >> 5) + it * 8;          // 0..31
      int n = (tid & 31) * 4;               // 0..124
      const bf16* src = W + (size_t)(kt + k) * Np + bn + n;
#pragma unroll
      for (int e = 0; e < 4; ++e) {
        bf16 v = (bf16)0.0f;
        if (bn + n + e < Np) v = src[e];
        Bs[buf][(n + e) * BK + k] = v;      // transpose into [n][k]
      }
    }
  };

  auto compute = [&](int buf) {
    const int kb   = (lane >> 4) * 8;       // lanes 16-31 hold K+8.. / K+24..
    const int mrow = lane & 15;
    ABFrag af[2], bf_[4];
#pragma unroll
    for (int i = 0; i < 2; ++i) {
      int base = (wm * 32 + i * 16 + mrow) * BK + kb;
      af[i].h[0] = *(const v8bf*)&As[buf][base];        // K 0..7 (+kb)
      af[i].h[1] = *(const v8bf*)&As[buf][base + 16];   // K 16..23 (+kb)
    }
#pragma unroll
    for (int j = 0; j < 4; ++j) {
      int base = (wn * 64 + j * 16 + mrow) * BK + kb;
      bf_[j].h[0] = *(const v8bf*)&Bs[buf][base];
      bf_[j].h[1] = *(const v8bf*)&Bs[buf][base + 16];
    }
#pragma unroll
    for (int i = 0; i < 2; ++i)
#pragma unroll
      for (int j = 0; j < 4; ++j)
        acc[i][j] = __builtin_amdgcn_wmma_f32_16x16x32_bf16(
            false, af[i].v, false, bf_[j].v, (short)0, acc[i][j], false, false);
  };

  const int nk = K / BK;
  tdmA(0, 0);
  loadB(0, 0);
  for (int t = 0; t < nk; ++t) {
    __builtin_amdgcn_s_wait_tensorcnt(0);   // TDM tile t landed (no-op for waves 1..7)
    __syncthreads();
    if (t + 1 < nk) {
      if (t + 2 < nk)   // speculative L2 prefetch of the W stream two tiles ahead
        __builtin_prefetch(W + (size_t)((t + 2) * BK + (tid >> 5)) * Np + bn, 0, 0);
      tdmA((t + 1) * BK, (t + 1) & 1);
      loadB((t + 1) * BK, (t + 1) & 1);
    }
    compute(t & 1);
  }

  // Epilogue: bias + activation + store.
  // C/D layout: lane 0-15 -> N=lane, rows r..r+7 in VGPR r; lane 16-31 -> rows 8+r.
#pragma unroll
  for (int i = 0; i < 2; ++i) {
#pragma unroll
    for (int j = 0; j < 4; ++j) {
      int col = bn + wn * 64 + j * 16 + (lane & 15);
      float bv = (col < Nbias) ? bias[col] : 0.0f;
      int rbase = bm + wm * 32 + i * 16 + ((lane >> 4) << 3);
#pragma unroll
      for (int r = 0; r < 8; ++r) {
        float v = acc[i][j][r] + bv;
        if (ACT == 0) {
          v = v > 0.0f ? v : 0.0f;
        } else {
          v = 1.0f / (1.0f + __expf(-v));
        }
        if (col < Np) out[(size_t)(rbase + r) * Np + col] = (OutT)v;
      }
    }
  }
}

// ---------------------------------------------------------------------------
// Host launch
// ---------------------------------------------------------------------------
extern "C" void kernel_launch(void* const* d_in, const int* in_sizes, int n_in,
                              void* d_out, int out_size, void* d_ws, size_t ws_size,
                              hipStream_t stream)
{
  (void)in_sizes; (void)n_in; (void)out_size; (void)ws_size;

  const float* x   = (const float*)d_in[0];
  const float* C1  = (const float*)d_in[1];
  const float* W1  = (const float*)d_in[2];
  const float* b1  = (const float*)d_in[3];
  const float* C2  = (const float*)d_in[4];
  const float* W2  = (const float*)d_in[5];
  const float* b2  = (const float*)d_in[6];
  const float* W3  = (const float*)d_in[7];
  const float* b3  = (const float*)d_in[8];
  const float* W4  = (const float*)d_in[9];
  const float* b4  = (const float*)d_in[10];
  const float* Wd1 = (const float*)d_in[11];
  const float* bd1 = (const float*)d_in[12];
  const float* Wd2 = (const float*)d_in[13];
  const float* bd2 = (const float*)d_in[14];
  const float* Wd3 = (const float*)d_in[15];
  const float* bd3 = (const float*)d_in[16];

  char* ws = (char*)d_ws;
  auto alloc = [&](size_t elems) {
    char* p = ws;
    ws += (elems * sizeof(bf16) + 255) & ~(size_t)255;
    return (bf16*)p;
  };

  bf16* xb   = alloc((size_t)Bdim * Sdim);
  bf16* W1m  = alloc((size_t)Sdim * H1P);
  bf16* h1   = alloc((size_t)Bdim * H1P);
  bf16* W2m  = alloc((size_t)H1P * H2P);
  bf16* h2   = alloc((size_t)Bdim * H2P);
  bf16* W3p  = alloc((size_t)H2P * D4);
  bf16* h3   = alloc((size_t)Bdim * D4);
  bf16* W4b  = alloc((size_t)D4 * LAT);
  bf16* zb   = alloc((size_t)Bdim * LAT);
  bf16* Wd1b = alloc((size_t)LAT * DD1);
  bf16* dd1  = alloc((size_t)Bdim * DD1);
  bf16* Wd2b = alloc((size_t)DD1 * DD2);
  bf16* dd2  = alloc((size_t)Bdim * DD2);
  bf16* Wd3b = alloc((size_t)DD2 * Sdim);

  auto cvt = [&](const float* src, const float* mask, bf16* dst,
                 int rows, int cols, int rowsP, int colsP) {
    size_t total = (size_t)rowsP * colsP;
    cvt_pad_bf16<<<dim3((unsigned)((total + 255) / 256)), 256, 0, stream>>>(
        src, mask, dst, rows, cols, rowsP, colsP);
  };

  // Conversions / mask folding / padding
  cvt(x,   nullptr, xb,   Bdim, Sdim, Bdim, Sdim);
  cvt(W1,  C1,      W1m,  Sdim, H1,   Sdim, H1P);
  cvt(W2,  C2,      W2m,  H1,   H2,   H1P,  H2P);
  cvt(W3,  nullptr, W3p,  H2,   D4,   H2P,  D4);
  cvt(W4,  nullptr, W4b,  D4,   LAT,  D4,   LAT);
  cvt(Wd1, nullptr, Wd1b, LAT,  DD1,  LAT,  DD1);
  cvt(Wd2, nullptr, Wd2b, DD1,  DD2,  DD1,  DD2);
  cvt(Wd3, nullptr, Wd3b, DD2,  Sdim, DD2,  Sdim);

  auto gemm_relu = [&](const bf16* A, const bf16* Wt, const float* bias,
                       bf16* out, int M, int K, int Np, int Nbias) {
    dim3 g(M / BM, (Np + BN - 1) / BN);
    gemm_bias_act<0, bf16><<<g, 256, 0, stream>>>(A, Wt, bias, out, M, K, Np, Nbias);
  };

  // Encoder
  gemm_relu(xb, W1m, b1, h1, Bdim, Sdim, H1P, H1);
  gemm_relu(h1, W2m, b2, h2, Bdim, H1P,  H2P, H2);
  gemm_relu(h2, W3p, b3, h3, Bdim, H2P,  D4,  D4);
  gemm_relu(h3, W4b, b4, zb, Bdim, D4,   LAT, LAT);
  // Decoder
  gemm_relu(zb,  Wd1b, bd1, dd1, Bdim, LAT, DD1, DD1);
  gemm_relu(dd1, Wd2b, bd2, dd2, Bdim, DD1, DD2, DD2);
  // Final layer: sigmoid, fp32 output
  {
    dim3 g(Bdim / BM, (Sdim + BN - 1) / BN);
    gemm_bias_act<1, float><<<g, 256, 0, stream>>>(
        dd2, Wd3b, bd3, (float*)d_out, Bdim, DD2, Sdim, Sdim);
  }
}
